// TransformerBlock_48361331753102
// MI455X (gfx1250) — matrix operand order repaired and checked by reading, not compiled, not run on hardware
//
#include <hip/hip_runtime.h>
#include <hip/hip_bf16.h>
#include <stdint.h>

// ---------------------------------------------------------------------------
// Types for CDNA5 WMMA (wave32). Storage of bf16 data is uint16_t; the WMMA
// operand vectors are built by bit-reinterpretation through a union.
// ---------------------------------------------------------------------------
typedef __attribute__((ext_vector_type(16))) __bf16 v16bf;
typedef __attribute__((ext_vector_type(8)))  float  v8f;

union FragBF16 {
    v16bf v;
    uint4 q[2];
};

// Async global->LDS path (CDNA5 GLOBAL_LOAD_ASYNC_TO_LDS_B128, ASYNCcnt).
// Guarded so the file compiles even if the builtin is absent on this toolchain.
#if defined(__gfx1250__) && __has_builtin(__builtin_amdgcn_global_load_async_to_lds_b128)
#define USE_ASYNC_LDS 1
#else
#define USE_ASYNC_LDS 0
#endif

#if USE_ASYNC_LDS
// Builtin prototype (from hipcc diagnostic): arg0 is a pointer to
// 'int __attribute__((vector_size(16)))' in the global address space, arg1 the
// LDS-side pointer, then immediate offset and cache-policy ints.
typedef int v4i_vs __attribute__((vector_size(16)));
typedef __attribute__((address_space(1))) v4i_vs* as1_v4i;
typedef __attribute__((address_space(3))) v4i_vs* as3_v4i;

static __device__ __forceinline__ void async_ld16(const uint16_t* g, const uint16_t* l) {
    // LDS address = low 32 bits of the generic pointer (flat->LDS truncation).
    __builtin_amdgcn_global_load_async_to_lds_b128(
        (as1_v4i)(uintptr_t)g,
        (as3_v4i)(uint32_t)(uintptr_t)l,
        /*offset=*/0, /*cpol=*/0);
}
static __device__ __forceinline__ void wait_async0() {
#if __has_builtin(__builtin_amdgcn_s_wait_asynccnt)
    __builtin_amdgcn_s_wait_asynccnt(0);
#else
    asm volatile("s_wait_asynccnt 0x0" ::: "memory");
#endif
}
#endif

static __device__ __forceinline__ uint16_t f32_to_bf16_bits(float f) {
    union { float f; uint32_t u; } in;
    in.f = f;
    uint32_t u = in.u;
    // round-to-nearest-even
    uint32_t r = (u + 0x7fffu + ((u >> 16) & 1u)) >> 16;
    return (uint16_t)r;
}

// ---------------------------------------------------------------------------
// Elementwise f32 -> bf16 convert
// ---------------------------------------------------------------------------
__global__ void k_cvt_bf16(const float* __restrict__ in, uint16_t* __restrict__ out, int n) {
    int i = blockIdx.x * blockDim.x + threadIdx.x;
    if (i < n) out[i] = f32_to_bf16_bits(in[i]);
}

// in: (K x N) f32 row-major  ->  out: (N x K) bf16 row-major (transpose)
__global__ void k_cvt_transpose_bf16(const float* __restrict__ in, uint16_t* __restrict__ out,
                                     int K, int N) {
    int i = blockIdx.x * blockDim.x + threadIdx.x;
    if (i < K * N) {
        int k = i / N;
        int n = i - k * N;
        out[(size_t)n * K + k] = f32_to_bf16_bits(in[i]);
    }
}

// ---------------------------------------------------------------------------
// Blocked GEMM: C(M,N) = scale * (A(M,Kd) @ Bt(N,Kd)^T) + bias, optional ReLU.
// A, Bt: bf16 row-major (Bt holds B transposed: element (n,k)).
//
// Workgroup = 256 threads = 8 waves computing a 128x128 C macro-tile.
// Waves arranged 4 (M) x 2 (N); each wave owns a 32x64 register tile
// = 2x4 accumulators of v_wmma_f32_16x16x32_bf16.
//
// Per 32-k step the workgroup stages A(128x32) and B(128x32) bf16 tiles in
// LDS. On toolchains exposing GLOBAL_LOAD_ASYNC_TO_LDS_B128 the staging is
// double-buffered and asynchronous (ASYNCcnt + s_wait_asynccnt): the next
// k-step's tiles stream into LDS while the current step's WMMAs execute.
// Otherwise falls back to load->ds_store staging.
// LDS row stride padded to 40 halfs (80B) to spread banks.
//
// Fragment layouts per CDNA5 ISA 7.12.2 (wave32, 16-bit data):
//   A 16x32: lane r=l&15 owns row r; group g=l>>4: halves 0..7 = K[g*8..g*8+7],
//            halves 8..15 = K[16+g*8 .. 16+g*8+7]
//   B 32x16: lane r owns column r; group g: halves 0..7 = K[g*8..g*8+7],
//            halves 8..15 = K[16+g*8 .. 16+g*8+7]
//   C 16x16 f32: lane r owns column r; vgpr j holds row (8*g + j)
// ---------------------------------------------------------------------------
template <bool OUT_BF16, bool STORE_T, bool RELU>
__global__ __launch_bounds__(256) void k_gemm_bf16_wmma(
        const uint16_t* __restrict__ A, int lda,
        const uint16_t* __restrict__ Bt, int ldb,
        void* __restrict__ C, int ldc,
        int Kd, float scale,
        const float* __restrict__ bias) {
    constexpr int LS = 40;  // LDS row stride in halfs (80B, padded)
#if USE_ASYNC_LDS
    __shared__ __align__(16) uint16_t Ash[2][128 * LS];
    __shared__ __align__(16) uint16_t Bsh[2][128 * LS];
#else
    __shared__ __align__(16) uint16_t Ash[1][128 * LS];
    __shared__ __align__(16) uint16_t Bsh[1][128 * LS];
#endif

    const int tid  = threadIdx.x;       // 0..255
    const int lane = tid & 31;
    const int wave = tid >> 5;          // 0..7
    const int wm   = wave & 3;          // wave row   (4 along M)
    const int wn   = wave >> 2;         // wave col   (2 along N)
    const int g    = lane >> 4;
    const int r    = lane & 15;

    const int m0 = blockIdx.y * 128;
    const int n0 = blockIdx.x * 128;

    // cooperative staging: each thread moves 16 halfs (32B) of A and of B
    const int ldRow = tid >> 1;          // 0..127
    const int ldCol = (tid & 1) * 16;    // 0 or 16
    const int ldOff = ldRow * LS + ldCol;

    const uint16_t* aSrc = A  + (size_t)(m0 + ldRow) * lda + ldCol;
    const uint16_t* bSrc = Bt + (size_t)(n0 + ldRow) * ldb + ldCol;

    v8f acc[2][4] = {};

#if USE_ASYNC_LDS
    // ---- double-buffered async pipeline ----
    {
        // prologue: stage k-step 0 into buffer 0
        async_ld16(aSrc, &Ash[0][ldOff]);
        async_ld16(aSrc + 8, &Ash[0][ldOff + 8]);
        async_ld16(bSrc, &Bsh[0][ldOff]);
        async_ld16(bSrc + 8, &Bsh[0][ldOff + 8]);
        wait_async0();
        __syncthreads();
    }
    int buf = 0;
    for (int kb = 0; kb < Kd; kb += 32) {
        if (kb + 32 < Kd) {
            const int nxt = buf ^ 1;
            async_ld16(aSrc + kb + 32, &Ash[nxt][ldOff]);
            async_ld16(aSrc + kb + 40, &Ash[nxt][ldOff + 8]);
            async_ld16(bSrc + kb + 32, &Bsh[nxt][ldOff]);
            async_ld16(bSrc + kb + 40, &Bsh[nxt][ldOff + 8]);
        }

        FragBF16 af[2], bf[4];
#pragma unroll
        for (int mi = 0; mi < 2; ++mi) {
            const uint16_t* p = &Ash[buf][(wm * 32 + mi * 16 + r) * LS];
            af[mi].q[0] = *(const uint4*)(p + g * 8);
            af[mi].q[1] = *(const uint4*)(p + 16 + g * 8);
        }
#pragma unroll
        for (int ni = 0; ni < 4; ++ni) {
            const uint16_t* p = &Bsh[buf][(wn * 64 + ni * 16 + r) * LS];
            bf[ni].q[0] = *(const uint4*)(p + g * 8);
            bf[ni].q[1] = *(const uint4*)(p + 16 + g * 8);
        }
#pragma unroll
        for (int mi = 0; mi < 2; ++mi) {
#pragma unroll
            for (int ni = 0; ni < 4; ++ni) {
                acc[mi][ni] = __builtin_amdgcn_wmma_f32_16x16x32_bf16(
                    false, af[mi].v, false, bf[ni].v,
                    (short)0, acc[mi][ni], false, false);
            }
        }

        wait_async0();     // next buffer fully landed in LDS
        __syncthreads();   // all waves done reading current buffer
        buf ^= 1;
    }
#else
    // ---- fallback: synchronous load -> ds_store staging ----
    for (int kb = 0; kb < Kd; kb += 32) {
        const uint4 a0 = *(const uint4*)(aSrc + kb);
        const uint4 a1 = *(const uint4*)(aSrc + kb + 8);
        const uint4 b0 = *(const uint4*)(bSrc + kb);
        const uint4 b1 = *(const uint4*)(bSrc + kb + 8);
        __syncthreads();
        *(uint4*)(&Ash[0][ldOff])     = a0;
        *(uint4*)(&Ash[0][ldOff + 8]) = a1;
        *(uint4*)(&Bsh[0][ldOff])     = b0;
        *(uint4*)(&Bsh[0][ldOff + 8]) = b1;
        __syncthreads();

        FragBF16 af[2], bf[4];
#pragma unroll
        for (int mi = 0; mi < 2; ++mi) {
            const uint16_t* p = &Ash[0][(wm * 32 + mi * 16 + r) * LS];
            af[mi].q[0] = *(const uint4*)(p + g * 8);
            af[mi].q[1] = *(const uint4*)(p + 16 + g * 8);
        }
#pragma unroll
        for (int ni = 0; ni < 4; ++ni) {
            const uint16_t* p = &Bsh[0][(wn * 64 + ni * 16 + r) * LS];
            bf[ni].q[0] = *(const uint4*)(p + g * 8);
            bf[ni].q[1] = *(const uint4*)(p + 16 + g * 8);
        }
#pragma unroll
        for (int mi = 0; mi < 2; ++mi) {
#pragma unroll
            for (int ni = 0; ni < 4; ++ni) {
                acc[mi][ni] = __builtin_amdgcn_wmma_f32_16x16x32_bf16(
                    false, af[mi].v, false, bf[ni].v,
                    (short)0, acc[mi][ni], false, false);
            }
        }
    }
#endif

    // epilogue
#pragma unroll
    for (int ni = 0; ni < 4; ++ni) {
        const int n = n0 + wn * 64 + ni * 16 + r;   // column owned by this lane
        const float bv = bias ? bias[n] : 0.0f;
#pragma unroll
        for (int mi = 0; mi < 2; ++mi) {
#pragma unroll
            for (int j = 0; j < 8; ++j) {
                const int m = m0 + wm * 32 + mi * 16 + g * 8 + j;
                float v = acc[mi][ni][j] * scale + bv;
                if (RELU) v = fmaxf(v, 0.0f);
                const size_t idx = STORE_T ? ((size_t)n * ldc + m) : ((size_t)m * ldc + n);
                if (OUT_BF16) ((uint16_t*)C)[idx] = f32_to_bf16_bits(v);
                else          ((float*)C)[idx] = v;
            }
        }
    }
}

// ---------------------------------------------------------------------------
// Row softmax: S (rows x n) f32 -> P bf16. One 256-thread block per row.
// ---------------------------------------------------------------------------
__global__ void k_softmax_bf16(const float* __restrict__ S, uint16_t* __restrict__ P, int n) {
    __shared__ float red[256];
    const int row = blockIdx.x;
    const int t = threadIdx.x;
    const float* s = S + (size_t)row * n;
    uint16_t* p = P + (size_t)row * n;

    float mx = -INFINITY;
    for (int i = t; i < n; i += 256) mx = fmaxf(mx, s[i]);
    red[t] = mx; __syncthreads();
    for (int off = 128; off > 0; off >>= 1) {
        if (t < off) red[t] = fmaxf(red[t], red[t + off]);
        __syncthreads();
    }
    mx = red[0]; __syncthreads();

    float sum = 0.0f;
    for (int i = t; i < n; i += 256) sum += __expf(s[i] - mx);
    red[t] = sum; __syncthreads();
    for (int off = 128; off > 0; off >>= 1) {
        if (t < off) red[t] += red[t + off];
        __syncthreads();
    }
    const float inv = 1.0f / red[0];

    for (int i = t; i < n; i += 256) p[i] = f32_to_bf16_bits(__expf(s[i] - mx) * inv);
}

// ---------------------------------------------------------------------------
// out = LayerNorm(X + Y) * gamma + beta  (row width E = 256 == blockDim.x)
// Writes f32 output and (optionally) a bf16 copy for the next GEMM's A side.
// ---------------------------------------------------------------------------
__global__ void k_add_layernorm(const float* __restrict__ X, const float* __restrict__ Y,
                                const float* __restrict__ gamma, const float* __restrict__ beta,
                                float* __restrict__ out_f32, uint16_t* __restrict__ out_bf16) {
    __shared__ float red[256];
    const int row = blockIdx.x;
    const int t = threadIdx.x;
    const size_t base = (size_t)row * 256;

    const float v = X[base + t] + Y[base + t];
    red[t] = v; __syncthreads();
    for (int off = 128; off > 0; off >>= 1) {
        if (t < off) red[t] += red[t + off];
        __syncthreads();
    }
    const float mu = red[0] * (1.0f / 256.0f);
    __syncthreads();

    const float d = v - mu;
    red[t] = d * d; __syncthreads();
    for (int off = 128; off > 0; off >>= 1) {
        if (t < off) red[t] += red[t + off];
        __syncthreads();
    }
    const float var = red[0] * (1.0f / 256.0f);

    const float res = d * rsqrtf(var + 1e-6f) * gamma[t] + beta[t];
    out_f32[base + t] = res;
    if (out_bf16) out_bf16[base + t] = f32_to_bf16_bits(res);
}

// ---------------------------------------------------------------------------
// Host orchestration
// ---------------------------------------------------------------------------
extern "C" void kernel_launch(void* const* d_in, const int* in_sizes, int n_in,
                              void* d_out, int out_size, void* d_ws, size_t ws_size,
                              hipStream_t stream) {
    constexpr int B = 2, S = 2048, E = 256, H = 8, DH = 256;
    constexpr int INNER = H * DH;   // 2048
    constexpr int DFF = 1024;
    constexpr int BS = B * S;       // 4096

    const float* x     = (const float*)d_in[0];
    const float* W_q   = (const float*)d_in[1];
    const float* W_k   = (const float*)d_in[2];
    const float* W_v   = (const float*)d_in[3];
    const float* W_o   = (const float*)d_in[4];
    const float* W1    = (const float*)d_in[5];
    const float* b1    = (const float*)d_in[6];
    const float* W2    = (const float*)d_in[7];
    const float* b2    = (const float*)d_in[8];
    const float* ln1_g = (const float*)d_in[9];
    const float* ln1_b = (const float*)d_in[10];
    const float* ln2_g = (const float*)d_in[11];
    const float* ln2_b = (const float*)d_in[12];

    // Workspace carve-up (256B aligned slices)
    char* wp = (char*)d_ws;
    auto carve = [&](size_t bytes) -> char* {
        char* p = wp;
        wp += (bytes + 255) & ~(size_t)255;
        return p;
    };
    uint16_t* xh   = (uint16_t*)carve((size_t)BS * E * 2);          // x in bf16
    uint16_t* Wqt  = (uint16_t*)carve((size_t)INNER * E * 2);       // W_q^T
    uint16_t* Wkt  = (uint16_t*)carve((size_t)INNER * E * 2);
    uint16_t* Wvt  = (uint16_t*)carve((size_t)INNER * E * 2);
    uint16_t* Wot  = (uint16_t*)carve((size_t)E * INNER * 2);       // W_o^T
    uint16_t* W1t  = (uint16_t*)carve((size_t)DFF * E * 2);
    uint16_t* W2t  = (uint16_t*)carve((size_t)E * DFF * 2);
    uint16_t* Qh   = (uint16_t*)carve((size_t)BS * INNER * 2);      // Q bf16 [B,S,INNER]
    uint16_t* Kh   = (uint16_t*)carve((size_t)BS * INNER * 2);      // K bf16
    uint16_t* Vt   = (uint16_t*)carve((size_t)B * H * DH * S * 2);  // V^T bf16 [B,H,DH,S]
    float*    scr  = (float*)   carve((size_t)S * S * 4);           // scores (reused per b,h)
    uint16_t* attn = (uint16_t*)carve((size_t)S * S * 2);           // softmax bf16 (reused)
    uint16_t* oall = (uint16_t*)carve((size_t)BS * INNER * 2);      // attn@V, heads merged
    float*    mha  = (float*)   carve((size_t)BS * E * 4);          // o @ W_o
    float*    hf   = (float*)   carve((size_t)BS * E * 4);          // LN1 out f32
    uint16_t* hb   = (uint16_t*)carve((size_t)BS * E * 2);          // LN1 out bf16
    uint16_t* f1   = (uint16_t*)carve((size_t)BS * DFF * 2);        // relu(h@W1+b1) bf16
    float*    f2   = (float*)   carve((size_t)BS * E * 4);          // ffn out f32
    (void)ws_size; (void)in_sizes; (void)n_in; (void)out_size;

    auto cdiv = [](int a, int b) { return (a + b - 1) / b; };

    // --- Stage 0: precision conversion + weight transposes ---
    k_cvt_bf16<<<cdiv(BS * E, 256), 256, 0, stream>>>(x, xh, BS * E);
    k_cvt_transpose_bf16<<<cdiv(E * INNER, 256), 256, 0, stream>>>(W_q, Wqt, E, INNER);
    k_cvt_transpose_bf16<<<cdiv(E * INNER, 256), 256, 0, stream>>>(W_k, Wkt, E, INNER);
    k_cvt_transpose_bf16<<<cdiv(E * INNER, 256), 256, 0, stream>>>(W_v, Wvt, E, INNER);
    k_cvt_transpose_bf16<<<cdiv(INNER * E, 256), 256, 0, stream>>>(W_o, Wot, INNER, E);
    k_cvt_transpose_bf16<<<cdiv(E * DFF, 256), 256, 0, stream>>>(W1, W1t, E, DFF);
    k_cvt_transpose_bf16<<<cdiv(DFF * E, 256), 256, 0, stream>>>(W2, W2t, DFF, E);

    // --- Stage 1: Q/K/V projections (x @ W) ---
    // Q, K: (BS x E) @ (E x INNER) -> bf16 row-major
    k_gemm_bf16_wmma<true, false, false><<<dim3(INNER / 128, BS / 128), 256, 0, stream>>>(
        xh, E, Wqt, E, Qh, INNER, E, 1.0f, nullptr);
    k_gemm_bf16_wmma<true, false, false><<<dim3(INNER / 128, BS / 128), 256, 0, stream>>>(
        xh, E, Wkt, E, Kh, INNER, E, 1.0f, nullptr);
    // V: per-batch, stored transposed -> Vt[b][h*DH+d][i]
    for (int b = 0; b < B; ++b) {
        k_gemm_bf16_wmma<true, true, false><<<dim3(INNER / 128, S / 128), 256, 0, stream>>>(
            xh + (size_t)b * S * E, E, Wvt, E,
            Vt + (size_t)b * H * DH * S, S, E, 1.0f, nullptr);
    }

    // --- Stage 2: attention per (batch, head) ---
    const float inv_sqrt_dh = 0.0625f;  // 1/sqrt(256)
    for (int b = 0; b < B; ++b) {
        for (int h = 0; h < H; ++h) {
            const uint16_t* Qbh = Qh + (size_t)b * S * INNER + (size_t)h * DH;
            const uint16_t* Kbh = Kh + (size_t)b * S * INNER + (size_t)h * DH;
            const uint16_t* Vbh = Vt + ((size_t)b * H + h) * DH * S;  // (DH x S)
            uint16_t* Obh = oall + (size_t)b * S * INNER + (size_t)h * DH;

            // scores = Q @ K^T / sqrt(DH):  (S x DH) x (S x DH)^T -> f32 (S x S)
            k_gemm_bf16_wmma<false, false, false><<<dim3(S / 128, S / 128), 256, 0, stream>>>(
                Qbh, INNER, Kbh, INNER, scr, S, DH, inv_sqrt_dh, nullptr);
            // softmax rows -> bf16
            k_softmax_bf16<<<S, 256, 0, stream>>>(scr, attn, S);
            // o = attn @ V: (S x S) x (S x DH); Bt = V^T (DH x S)
            k_gemm_bf16_wmma<true, false, false><<<dim3(DH / 128, S / 128), 256, 0, stream>>>(
                attn, S, Vbh, S, Obh, INNER, S, 1.0f, nullptr);
        }
    }

    // --- Stage 3: output projection + LN1 ---
    k_gemm_bf16_wmma<false, false, false><<<dim3(E / 128, BS / 128), 256, 0, stream>>>(
        oall, INNER, Wot, INNER, mha, E, INNER, 1.0f, nullptr);
    k_add_layernorm<<<BS, 256, 0, stream>>>(x, mha, ln1_g, ln1_b, hf, hb);

    // --- Stage 4: FFN + LN2 ---
    k_gemm_bf16_wmma<true, false, true><<<dim3(DFF / 128, BS / 128), 256, 0, stream>>>(
        hb, E, W1t, E, f1, DFF, E, 1.0f, b1);
    k_gemm_bf16_wmma<false, false, false><<<dim3(E / 128, BS / 128), 256, 0, stream>>>(
        f1, DFF, W2t, DFF, f2, E, DFF, 1.0f, b2);
    k_add_layernorm<<<BS, 256, 0, stream>>>(hf, f2, ln2_g, ln2_b, (float*)d_out, nullptr);
}
